// Mamba2AD_19035295056417
// MI455X (gfx1250) — matrix operand order repaired and checked by reading, not compiled, not run on hardware
//
#include <hip/hip_runtime.h>
#include <cstdint>
#include <cstddef>

// ---------------- problem constants ----------------
static constexpr int BATCH    = 16;
static constexpr int D_MODEL  = 256;
static constexpr int D_INNER  = 512;
static constexpr int HEADDIM  = 64;
static constexpr int NHEADS   = 8;
static constexpr int D_STATE  = 64;
static constexpr int CONV_DIM = 640;   // D_INNER + 2*D_STATE
static constexpr int D_IN_PROJ= 1160;  // 2*512 + 2*64 + 8
static constexpr int NPROJ_PAD= 1216;  // padded to 19*64 for guard-free GEMM
static constexpr int HH       = 64;
static constexpr int WW       = 64;
static constexpr int LL       = HH * WW;        // 4096
static constexpr int MROWS    = BATCH * LL;     // 65536

// ---------------- WMMA types ----------------
typedef __attribute__((ext_vector_type(16))) __bf16 v16bf;
typedef __attribute__((ext_vector_type(8)))  float  v8f;

union BF16Frag { v16bf v; unsigned short s[16]; uint32_t u[8]; uint4 q[2]; };
union F8       { v8f v; float f[8]; };

__device__ __forceinline__ unsigned short f2bf(float f) {
    union { float f; uint32_t u; } x; x.f = f;
    uint32_t r = x.u + 0x7FFFu + ((x.u >> 16) & 1u);
    return (unsigned short)(r >> 16);
}
__device__ __forceinline__ float bf2f(unsigned short h) {
    union { uint32_t u; float f; } x; x.u = ((uint32_t)h) << 16;
    return x.f;
}
__device__ __forceinline__ float silu_f(float x) { return x / (1.0f + expf(-x)); }
__device__ __forceinline__ float softplus_f(float x) {
    return (x > 20.0f) ? x : log1pf(expf(x));
}
__device__ __forceinline__ v8f wmma_bf16(const BF16Frag& a, const BF16Frag& b, v8f c) {
    return __builtin_amdgcn_wmma_f32_16x16x32_bf16(false, a.v, false, b.v,
                                                   (short)0, c, false, false);
}
// A-pair base K offset for VGPR j (16-bit A 16x32 layout)
__device__ __forceinline__ int a_kbase(int j, int hl) {
    return ((j < 4) ? (2 * j) : (16 + 2 * (j - 4))) + 8 * hl;
}

// ---- CDNA5 async global->LDS copy (ASYNCcnt-tracked, ISA 07_vmem op 98) ----
__device__ __forceinline__ void async_copy_b128(const void* gsrc, void* ldst) {
    unsigned loff = (unsigned)(unsigned long long)ldst;  // low 32b of generic = LDS offset
    asm volatile("global_load_async_to_lds_b128 %0, %1, off"
                 :: "v"(loff), "v"(gsrc) : "memory");
}
__device__ __forceinline__ void wait_asynccnt0() {
    asm volatile("s_wait_asynccnt 0x0" ::: "memory");
}

// =====================================================================
// K0a: transpose/convert u (B, C, H, W) f32 -> X[m, c] bf16 (m = b*L + h*W + w)
// =====================================================================
__global__ __launch_bounds__(256) void k_transpose_u(const float* __restrict__ u,
                                                     unsigned short* __restrict__ X) {
    __shared__ float tile[32][33];
    int bh = blockIdx.x >> 1;
    int wt = blockIdx.x & 1;
    int ct = blockIdx.y;
    int b  = bh >> 6;
    int hh = bh & 63;
    int w0 = wt * 32, c0 = ct * 32;
    int tx = threadIdx.x, ty = threadIdx.y;
#pragma unroll
    for (int j = 0; j < 4; ++j) {
        int c = c0 + ty + j * 8;
        tile[ty + j * 8][tx] = u[(((size_t)b * 256 + c) * 64 + hh) * 64 + w0 + tx];
    }
    __syncthreads();
#pragma unroll
    for (int j = 0; j < 4; ++j) {
        int wl = ty + j * 8;
        size_t m = (size_t)b * LL + hh * 64 + w0 + wl;
        X[m * 256 + c0 + tx] = f2bf(tile[tx][wl]);
    }
}

// K0b: convert W_in -> bf16 zero-padded to 1216 rows; W_out -> bf16
__global__ __launch_bounds__(256) void k_convert_weights(const float* __restrict__ Win,
                                                         const float* __restrict__ Wout,
                                                         unsigned short* __restrict__ Winb,
                                                         unsigned short* __restrict__ Woutb) {
    int idx = blockIdx.x * 256 + threadIdx.x;
    const int n1p = NPROJ_PAD * D_MODEL;     // padded W_in extent
    const int n1  = D_IN_PROJ * D_MODEL;
    const int n2  = D_MODEL * D_INNER;
    if (idx < n1p) {
        Winb[idx] = (idx < n1) ? f2bf(Win[idx]) : (unsigned short)0;
    } else if (idx < n1p + n2) {
        Woutb[idx - n1p] = f2bf(Wout[idx - n1p]);
    }
}

// =====================================================================
// K1: in-proj GEMM: zxbcdt[m, n] = sum_k X[m,k] * Win[n,k]
//     M=65536, K=256, N=1160(pad 1216). Block: 8 waves -> 128 rows x 64 cols.
//     Weight tile (64 x 32 bf16) staged in LDS via async copies, double buffered.
// =====================================================================
__global__ __launch_bounds__(256) void k_inproj(const unsigned short* __restrict__ X,
                                                const unsigned short* __restrict__ Wb,
                                                unsigned short* __restrict__ zb,
                                                unsigned short* __restrict__ xBCpre,
                                                float* __restrict__ dtraw) {
    __shared__ unsigned short wtile[2][64 * 32];  // [buf][row n][k] 4KB each
    int wave = threadIdx.x >> 5;
    int lane = threadIdx.x & 31;
    int ln = lane & 15, hl = lane >> 4;
    int m0 = blockIdx.x * 128 + wave * 16;
    int n0 = blockIdx.y * 64;

    // async-stage one 64x32 weight tile: 256 threads x 16B
    int crow = threadIdx.x >> 2;          // 0..63
    int cch  = threadIdx.x & 3;           // 4 chunks of 8 bf16 per row
    auto stage = [&](int buf, int k0) {
        async_copy_b128(Wb + (size_t)(n0 + crow) * D_MODEL + k0 + cch * 8,
                        &wtile[buf][crow * 32 + cch * 8]);
    };

    F8 acc[4];
#pragma unroll
    for (int t = 0; t < 4; ++t)
#pragma unroll
        for (int i = 0; i < 8; ++i) acc[t].f[i] = 0.0f;

    stage(0, 0);
    for (int it = 0; it < D_MODEL / 32; ++it) {
        int k0 = it * 32;
        int buf = it & 1;
        wait_asynccnt0();
        __syncthreads();
        if (it + 1 < D_MODEL / 32) stage(buf ^ 1, k0 + 32);

        BF16Frag a;  // A row = m0+ln: two contiguous 16B spans per lane
        const unsigned short* arow = X + (size_t)(m0 + ln) * D_MODEL + k0;
        a.q[0] = *(const uint4*)(arow + 16 * hl);
        a.q[1] = *(const uint4*)(arow + 16 * hl + 8);
#pragma unroll
        for (int t = 0; t < 4; ++t) {
            BF16Frag bfr;
            const uint4* pl = (const uint4*)&wtile[buf][(t * 16 + ln) * 32 + 16 * hl];
            bfr.q[0] = pl[0];
            bfr.q[1] = pl[1];
            acc[t].v = wmma_bf16(a, bfr, acc[t].v);
        }
    }
#pragma unroll
    for (int t = 0; t < 4; ++t) {
        int n = n0 + t * 16 + ln;
        if (n >= D_IN_PROJ) continue;
#pragma unroll
        for (int r = 0; r < 8; ++r) {
            size_t m = (size_t)m0 + r + 8 * hl;
            float v = acc[t].f[r];
            if (n < D_INNER) {
                zb[m * D_INNER + n] = f2bf(v);
            } else if (n < D_INNER + CONV_DIM) {
                xBCpre[m * CONV_DIM + (n - D_INNER)] = f2bf(v);
            } else {
                dtraw[m * NHEADS + (n - D_INNER - CONV_DIM)] = v;
            }
        }
    }
}

// =====================================================================
// K2: depthwise 3x3 conv (SAME) + bias + SiLU, plus dt softplus and V = x*dA
// =====================================================================
__global__ __launch_bounds__(256) void k_conv(const unsigned short* __restrict__ xBCpre,
                                              const float* __restrict__ conv_w,
                                              const float* __restrict__ conv_b,
                                              const float* __restrict__ dtraw,
                                              const float* __restrict__ dt_bias,
                                              const float* __restrict__ A_log,
                                              unsigned short* __restrict__ xsAct,
                                              unsigned short* __restrict__ Vs,
                                              unsigned short* __restrict__ Bm,
                                              unsigned short* __restrict__ Cm) {
    int idx = blockIdx.x * 256 + threadIdx.x;
    int c = idx % CONV_DIM;
    int m = idx / CONV_DIM;
    int b = m >> 12;
    int l = m & (LL - 1);
    int hh = l >> 6, ww = l & 63;

    float acc = conv_b[c];
#pragma unroll
    for (int ky = 0; ky < 3; ++ky) {
        int ih = hh + ky - 1;
        if (ih < 0 || ih >= HH) continue;
#pragma unroll
        for (int kx = 0; kx < 3; ++kx) {
            int iw = ww + kx - 1;
            if (iw < 0 || iw >= WW) continue;
            size_t mm = (size_t)b * LL + ih * 64 + iw;
            acc += bf2f(xBCpre[mm * CONV_DIM + c]) * conv_w[c * 9 + ky * 3 + kx];
        }
    }
    float val = silu_f(acc);
    if (c < D_INNER) {
        int head = c >> 6;
        xsAct[(size_t)m * D_INNER + c] = f2bf(val);
        float dtv = softplus_f(dtraw[(size_t)m * NHEADS + head] + dt_bias[head]);
        float dA = dtv * expf(A_log[head]);   // dA = -(dt*A), A = -exp(A_log)
        Vs[(size_t)m * D_INNER + c] = f2bf(val * dA);
    } else if (c < D_INNER + D_STATE) {
        Bm[(size_t)m * D_STATE + (c - D_INNER)] = f2bf(val);
    } else {
        Cm[(size_t)m * D_STATE + (c - D_INNER - D_STATE)] = f2bf(val);
    }
}

// =====================================================================
// K3: KV[b,h,n,d] = sum_l Bm[b,l,n] * Vs[b,l,h,d].  64x64, K=4096.
// One block per (b,h); Bm/Vs 32-row tiles async-staged to LDS (double buffer);
// 8 waves cover the 4x4 grid of 16x16 tiles (2 each).
// =====================================================================
__global__ __launch_bounds__(256) void k_kv(const unsigned short* __restrict__ Bm,
                                            const unsigned short* __restrict__ Vs,
                                            unsigned short* __restrict__ KVb) {
    __shared__ unsigned short btile[2][32 * 64];  // [buf][l][n] 4KB
    __shared__ unsigned short vtile[2][32 * 64];  // [buf][l][d] 4KB
    int b = blockIdx.x >> 3;
    int h = blockIdx.x & 7;
    int wave = threadIdx.x >> 5;
    int lane = threadIdx.x & 31;
    int ln = lane & 15, hl = lane >> 4;
    int nt = wave & 3;
    int d0 = (wave >> 2) * 2;
    int d1 = d0 + 1;
    const size_t rowB = (size_t)b * LL;

    // stage: 2 x 4KB, 256 threads x 2 chunks of 16B
    int crow = threadIdx.x >> 3;     // 0..31 (l within tile)
    int cch  = threadIdx.x & 7;      // 8 chunks of 8 bf16 per 64-wide row
    auto stage = [&](int buf, int l0) {
        async_copy_b128(Bm + (rowB + l0 + crow) * D_STATE + cch * 8,
                        &btile[buf][crow * 64 + cch * 8]);
        async_copy_b128(Vs + (rowB + l0 + crow) * D_INNER + h * HEADDIM + cch * 8,
                        &vtile[buf][crow * 64 + cch * 8]);
    };

    F8 acc0, acc1;
#pragma unroll
    for (int i = 0; i < 8; ++i) { acc0.f[i] = 0.0f; acc1.f[i] = 0.0f; }

    stage(0, 0);
    for (int it = 0; it < LL / 32; ++it) {
        int buf = it & 1;
        wait_asynccnt0();
        __syncthreads();
        if (it + 1 < LL / 32) stage(buf ^ 1, (it + 1) * 32);

        BF16Frag a;           // A[m=n][k=l] = btile[l][n]
#pragma unroll
        for (int j = 0; j < 8; ++j) {
            int kb = a_kbase(j, hl);
            a.s[2 * j]     = btile[buf][kb * 64 + nt * 16 + ln];
            a.s[2 * j + 1] = btile[buf][(kb + 1) * 64 + nt * 16 + ln];
        }
        BF16Frag f0, f1;      // B[k=l][n=d] = vtile[l][d]
#pragma unroll
        for (int e = 0; e < 16; ++e) {
            int lrow = (e < 8) ? (e + 8 * hl) : (e + 8 + 8 * hl);
            f0.s[e] = vtile[buf][lrow * 64 + d0 * 16 + ln];
            f1.s[e] = vtile[buf][lrow * 64 + d1 * 16 + ln];
        }
        acc0.v = wmma_bf16(a, f0, acc0.v);
        acc1.v = wmma_bf16(a, f1, acc1.v);
    }
    size_t base = ((size_t)b * NHEADS + h) * D_STATE;
#pragma unroll
    for (int r = 0; r < 8; ++r) {
        int n = nt * 16 + r + 8 * hl;
        KVb[(base + n) * HEADDIM + d0 * 16 + ln] = f2bf(acc0.f[r]);
        KVb[(base + n) * HEADDIM + d1 * 16 + ln] = f2bf(acc1.f[r]);
    }
}

// =====================================================================
// K4: y = C*KV + xs*D; gate with silu(z); LayerNorm over 512; store bf16.
// Block: 16 rows; wave w == head h does 16x64 WMMA; LDS + wave32 shuffles for LN.
// =====================================================================
__global__ __launch_bounds__(256) void k_y_gate_ln(const unsigned short* __restrict__ Cm,
                                                   const unsigned short* __restrict__ KVb,
                                                   const unsigned short* __restrict__ xsAct,
                                                   const unsigned short* __restrict__ zb,
                                                   const float* __restrict__ Dv,
                                                   const float* __restrict__ ln_g,
                                                   const float* __restrict__ ln_b,
                                                   unsigned short* __restrict__ yb) {
    __shared__ float ytile[16][512];
    int row0 = blockIdx.x * 16;
    int b = row0 >> 12;
    int h = threadIdx.x >> 5;
    int lane = threadIdx.x & 31;
    int ln = lane & 15, hl = lane >> 4;

    F8 acc[4];
#pragma unroll
    for (int t = 0; t < 4; ++t)
#pragma unroll
        for (int i = 0; i < 8; ++i) acc[t].f[i] = 0.0f;

    size_t kvbase = ((size_t)b * NHEADS + h) * D_STATE;
#pragma unroll
    for (int ks = 0; ks < D_STATE; ks += 32) {
        BF16Frag a;          // A[m=l][k=n] = Cm[row, n]: two 16B spans per lane
        const unsigned short* arow = Cm + (size_t)(row0 + ln) * D_STATE + ks;
        a.q[0] = *(const uint4*)(arow + 8 * hl);
        a.q[1] = *(const uint4*)(arow + 16 + 8 * hl);
#pragma unroll
        for (int t = 0; t < 4; ++t) {
            BF16Frag f;      // B[k=n][d] = KV[b,h,n,d]
#pragma unroll
            for (int e = 0; e < 16; ++e) {
                int n = ks + ((e < 8) ? (e + 8 * hl) : (e + 8 + 8 * hl));
                f.s[e] = KVb[(kvbase + n) * HEADDIM + t * 16 + ln];
            }
            acc[t].v = wmma_bf16(a, f, acc[t].v);
        }
    }
    float Dh = Dv[h];
#pragma unroll
    for (int t = 0; t < 4; ++t) {
#pragma unroll
        for (int r = 0; r < 8; ++r) {
            int rl = r + 8 * hl;
            int c = h * HEADDIM + t * 16 + ln;
            size_t m = (size_t)row0 + rl;
            float v = acc[t].f[r] + bf2f(xsAct[m * D_INNER + c]) * Dh;
            float zv = bf2f(zb[m * D_INNER + c]);
            ytile[rl][c] = v * silu_f(zv);
        }
    }
    __syncthreads();

#pragma unroll
    for (int q = 0; q < 2; ++q) {
        int rr = h + q * 8;
        float s = 0.0f, s2 = 0.0f;
#pragma unroll
        for (int i = 0; i < 16; ++i) {
            float v = ytile[rr][lane + i * 32];
            s += v; s2 += v * v;
        }
#pragma unroll
        for (int off = 16; off > 0; off >>= 1) {
            s  += __shfl_xor(s,  off, 32);
            s2 += __shfl_xor(s2, off, 32);
        }
        float mu = s * (1.0f / 512.0f);
        float var = s2 * (1.0f / 512.0f) - mu * mu;
        float rstd = rsqrtf(var + 1e-5f);
        size_t m = (size_t)row0 + rr;
#pragma unroll
        for (int i = 0; i < 16; ++i) {
            int c = lane + i * 32;
            float v = (ytile[rr][c] - mu) * rstd * ln_g[c] + ln_b[c];
            yb[m * D_INNER + c] = f2bf(v);
        }
    }
}

// =====================================================================
// K5: out-proj GEMM: out[b, o, l] = sum_k y[m,k] * Wout[o,k]
//     M=65536, K=512, N=256; weight tile async-staged in LDS; NCHW f32 store.
// =====================================================================
__global__ __launch_bounds__(256) void k_outproj(const unsigned short* __restrict__ yb,
                                                 const unsigned short* __restrict__ Wb,
                                                 float* __restrict__ out) {
    __shared__ unsigned short wtile[2][64 * 32];
    int wave = threadIdx.x >> 5;
    int lane = threadIdx.x & 31;
    int ln = lane & 15, hl = lane >> 4;
    int m0 = blockIdx.x * 128 + wave * 16;
    int n0 = blockIdx.y * 64;

    int crow = threadIdx.x >> 2;
    int cch  = threadIdx.x & 3;
    auto stage = [&](int buf, int k0) {
        async_copy_b128(Wb + (size_t)(n0 + crow) * D_INNER + k0 + cch * 8,
                        &wtile[buf][crow * 32 + cch * 8]);
    };

    F8 acc[4];
#pragma unroll
    for (int t = 0; t < 4; ++t)
#pragma unroll
        for (int i = 0; i < 8; ++i) acc[t].f[i] = 0.0f;

    stage(0, 0);
    for (int it = 0; it < D_INNER / 32; ++it) {
        int k0 = it * 32;
        int buf = it & 1;
        wait_asynccnt0();
        __syncthreads();
        if (it + 1 < D_INNER / 32) stage(buf ^ 1, k0 + 32);

        BF16Frag a;
        const unsigned short* arow = yb + (size_t)(m0 + ln) * D_INNER + k0;
        a.q[0] = *(const uint4*)(arow + 16 * hl);
        a.q[1] = *(const uint4*)(arow + 16 * hl + 8);
#pragma unroll
        for (int t = 0; t < 4; ++t) {
            BF16Frag f;
            const uint4* pl = (const uint4*)&wtile[buf][(t * 16 + ln) * 32 + 16 * hl];
            f.q[0] = pl[0];
            f.q[1] = pl[1];
            acc[t].v = wmma_bf16(a, f, acc[t].v);
        }
    }
#pragma unroll
    for (int t = 0; t < 4; ++t) {
        int n = n0 + t * 16 + ln;
#pragma unroll
        for (int r = 0; r < 8; ++r) {
            int m = m0 + r + 8 * hl;
            int b = m >> 12;
            int l = m & (LL - 1);
            out[((size_t)(b * D_MODEL + n)) * LL + l] = acc[t].f[r];
        }
    }
}

// =====================================================================
extern "C" void kernel_launch(void* const* d_in, const int* in_sizes, int n_in,
                              void* d_out, int out_size, void* d_ws, size_t ws_size,
                              hipStream_t stream) {
    const float* u       = (const float*)d_in[0];
    const float* W_in    = (const float*)d_in[1];
    const float* conv_w  = (const float*)d_in[2];
    const float* conv_b  = (const float*)d_in[3];
    const float* dt_bias = (const float*)d_in[4];
    const float* A_log   = (const float*)d_in[5];
    const float* Dvec    = (const float*)d_in[6];
    const float* ln_g    = (const float*)d_in[7];
    const float* ln_b    = (const float*)d_in[8];
    const float* W_out   = (const float*)d_in[9];
    float* out = (float*)d_out;

    char* ws = (char*)d_ws;
    size_t off = 0;
    auto take = [&](size_t bytes) -> char* {
        char* p = ws + off;
        off = (off + bytes + 255) & ~(size_t)255;
        return p;
    };
    unsigned short* Xb     = (unsigned short*)take((size_t)MROWS * D_MODEL * 2);
    unsigned short* Winb   = (unsigned short*)take((size_t)NPROJ_PAD * D_MODEL * 2);
    unsigned short* Woutb  = (unsigned short*)take((size_t)D_MODEL * D_INNER * 2);
    unsigned short* zb     = (unsigned short*)take((size_t)MROWS * D_INNER * 2);
    unsigned short* xBCpre = (unsigned short*)take((size_t)MROWS * CONV_DIM * 2);
    float*          dtraw  = (float*)take((size_t)MROWS * NHEADS * 4);
    unsigned short* xsAct  = (unsigned short*)take((size_t)MROWS * D_INNER * 2);
    unsigned short* Vs     = (unsigned short*)take((size_t)MROWS * D_INNER * 2);
    unsigned short* Bmb    = (unsigned short*)take((size_t)MROWS * D_STATE * 2);
    unsigned short* Cmb    = (unsigned short*)take((size_t)MROWS * D_STATE * 2);
    unsigned short* KVb    = (unsigned short*)take((size_t)BATCH * NHEADS * D_STATE * HEADDIM * 2);
    unsigned short* yb     = (unsigned short*)take((size_t)MROWS * D_INNER * 2);

    k_transpose_u<<<dim3(BATCH * HH * 2, 8), dim3(32, 8), 0, stream>>>(u, Xb);
    {
        int total = NPROJ_PAD * D_MODEL + D_MODEL * D_INNER;
        k_convert_weights<<<(total + 255) / 256, 256, 0, stream>>>(W_in, W_out, Winb, Woutb);
    }
    k_inproj<<<dim3(MROWS / 128, NPROJ_PAD / 64), 256, 0, stream>>>(
        Xb, Winb, zb, xBCpre, dtraw);
    {
        long long total = (long long)MROWS * CONV_DIM;
        k_conv<<<(unsigned)(total / 256), 256, 0, stream>>>(
            xBCpre, conv_w, conv_b, dtraw, dt_bias, A_log, xsAct, Vs, Bmb, Cmb);
    }
    k_kv<<<BATCH * NHEADS, 256, 0, stream>>>(Bmb, Vs, KVb);
    k_y_gate_ln<<<MROWS / 16, 256, 0, stream>>>(Cmb, KVb, xsAct, zb, Dvec, ln_g, ln_b, yb);
    k_outproj<<<dim3(MROWS / 128, D_MODEL / 64), 256, 0, stream>>>(yb, Woutb, out);
}
